// NSQLTransformer_88175678587303
// MI455X (gfx1250) — matrix operand order repaired and checked by reading, not compiled, not run on hardware
//
#include <hip/hip_runtime.h>
#include <hip/hip_bf16.h>

// ---------------------------------------------------------------------------
// Model constants (from the reference)
// ---------------------------------------------------------------------------
#define SEQ    1024
#define DMODEL 1024
#define NHEAD  16
#define HDIM   64
#define FFDIM  4096
#define VOCAB  51200
#define NLAYER 8
#define ROTD   32

typedef __attribute__((ext_vector_type(16))) __bf16 v16bf;
typedef __attribute__((ext_vector_type(8)))  __bf16 v8bf;
typedef __attribute__((ext_vector_type(4)))  __bf16 v4bf;
typedef __attribute__((ext_vector_type(8)))  float  v8f;

__device__ __forceinline__ v16bf cat16(v8bf lo, v8bf hi) {
    return __builtin_shufflevector(lo, hi, 0,1,2,3,4,5,6,7,8,9,10,11,12,13,14,15);
}

__device__ __forceinline__ float gelu_tanh(float x) {
    float x3 = x * x * x;
    return 0.5f * x * (1.0f + tanhf(0.7978845608028654f * (x + 0.044715f * x3)));
}

// ---------------------------------------------------------------------------
// CDNA5 async global->LDS copy (16 bytes / lane) + ASYNCcnt wait.
// LDS address comes from a VGPR (low 32 bits of the flat shared pointer are the
// LDS offset: the LDS aperture base has zero low bits).
// ---------------------------------------------------------------------------
__device__ __forceinline__ void async_copy_b128(void* lds, const void* gaddr) {
    unsigned           la = (unsigned)(unsigned long long)lds;
    unsigned long long ga = (unsigned long long)gaddr;
    asm volatile("global_load_async_to_lds_b128 %0, %1, off"
                 :: "v"(la), "v"(ga) : "memory");
}
__device__ __forceinline__ void wait_async0() {
    asm volatile("s_wait_asynccnt 0x0" ::: "memory");
}

// ---------------------------------------------------------------------------
// Embedding gather: x[s][d] = emb[idx[s]][d]   (fp32 residual stream)
// ---------------------------------------------------------------------------
__global__ __launch_bounds__(256) void embed_kernel(const int* __restrict__ idx,
                                                    const float* __restrict__ emb,
                                                    float* __restrict__ x) {
    int s = blockIdx.x;
    int row = idx[s];
    #pragma unroll
    for (int i = 0; i < 4; ++i) {
        int d = threadIdx.x + i * 256;
        x[s * DMODEL + d] = emb[row * DMODEL + d];
    }
}

// ---------------------------------------------------------------------------
// LayerNorm over rows of length DMODEL; fp32 math, bf16 output (GEMM operand).
// ---------------------------------------------------------------------------
__global__ __launch_bounds__(256) void layernorm_kernel(const float* __restrict__ x,
                                                        const float* __restrict__ w,
                                                        const float* __restrict__ b,
                                                        __bf16* __restrict__ out) {
    __shared__ float red[256];
    __shared__ float s_mu, s_rstd;
    int s = blockIdx.x;
    int tid = threadIdx.x;
    const float* row = x + s * DMODEL;

    float v0 = row[tid], v1 = row[tid + 256], v2 = row[tid + 512], v3 = row[tid + 768];
    red[tid] = v0 + v1 + v2 + v3;
    __syncthreads();
    for (int off = 128; off > 0; off >>= 1) {
        if (tid < off) red[tid] += red[tid + off];
        __syncthreads();
    }
    if (tid == 0) s_mu = red[0] * (1.0f / DMODEL);
    __syncthreads();
    float mu = s_mu;
    float d0 = v0 - mu, d1 = v1 - mu, d2 = v2 - mu, d3 = v3 - mu;
    red[tid] = d0 * d0 + d1 * d1 + d2 * d2 + d3 * d3;
    __syncthreads();
    for (int off = 128; off > 0; off >>= 1) {
        if (tid < off) red[tid] += red[tid + off];
        __syncthreads();
    }
    if (tid == 0) s_rstd = rsqrtf(red[0] * (1.0f / DMODEL) + 1e-5f);
    __syncthreads();
    float rstd = s_rstd;
    out[s * DMODEL + tid]       = (__bf16)(d0 * rstd * w[tid]       + b[tid]);
    out[s * DMODEL + tid + 256] = (__bf16)(d1 * rstd * w[tid + 256] + b[tid + 256]);
    out[s * DMODEL + tid + 512] = (__bf16)(d2 * rstd * w[tid + 512] + b[tid + 512]);
    out[s * DMODEL + tid + 768] = (__bf16)(d3 * rstd * w[tid + 768] + b[tid + 768]);
}

// ---------------------------------------------------------------------------
// WMMA GEMM:  C[M,N] = act( A[M,K](bf16) @ W[N,K]^T(fp32) + bias ) (+res0+res1)
// Block tile 256(M) x 64(N), 8 waves, wave = 32x64 (8 f32 accumulators).
// Double-buffered (ping-pong LDS) K pipeline:
//   - A tile for step i+1 streams in via ASYNCcnt-tracked global->LDS copies
//   - W slab for step i+1 sits in registers (global_load_b128) across the
//     WMMAs of step i, then is converted fp32->bf16 and ds-stored K-major
//   - one barrier per K step
// ---------------------------------------------------------------------------
template <bool GELU>
__global__ __launch_bounds__(256) void gemm_wmma(const __bf16* __restrict__ A,
                                                 const float* __restrict__ W,
                                                 const float* __restrict__ bias,
                                                 const float* __restrict__ res0,
                                                 const float* __restrict__ res1,
                                                 float* __restrict__ Cf,
                                                 __bf16* __restrict__ Cb,
                                                 int M, int N, int K) {
    __shared__ __bf16 As[2][256][32];  // 2 x 16 KB  [m][k]
    __shared__ __bf16 Bt[2][32][64];   // 2 x  4 KB  [k][n]

    const int tid  = threadIdx.x;
    const int lane = tid & 31;
    const int wave = tid >> 5;
    const int half = lane >> 4;
    const int l15  = lane & 15;
    const int koff = half ? 8 : 0;

    const int mb = blockIdx.y * 256;
    const int nb = blockIdx.x * 64;

    v8f acc[2][4];
    #pragma unroll
    for (int u = 0; u < 2; ++u)
        #pragma unroll
        for (int t = 0; t < 4; ++t)
            #pragma unroll
            for (int e = 0; e < 8; ++e) acc[u][t][e] = 0.0f;

    float4 wreg[2];

    // ---- staging helpers ----
    auto stageA = [&](int ks, int buf) {
        #pragma unroll
        for (int i = 0; i < 4; ++i) {
            int lin = tid + i * 256;
            int r = lin >> 2;
            int c = (lin & 3) * 8;
            async_copy_b128(&As[buf][r][c], &A[(size_t)(mb + r) * K + ks * 32 + c]);
        }
    };
    auto loadW = [&](int ks) {
        #pragma unroll
        for (int i = 0; i < 2; ++i) {
            int lin = tid + i * 256;
            int n = lin >> 3;
            int c = (lin & 7) * 4;
            wreg[i] = *(const float4*)&W[(size_t)(nb + n) * K + ks * 32 + c];
            if ((ks + 1) * 32 < K)
                __builtin_prefetch(&W[(size_t)(nb + n) * K + (ks + 1) * 32 + c], 0, 1);
        }
    };
    auto storeW = [&](int buf) {
        #pragma unroll
        for (int i = 0; i < 2; ++i) {
            int lin = tid + i * 256;
            int n = lin >> 3;
            int c = (lin & 7) * 4;
            Bt[buf][c + 0][n] = (__bf16)wreg[i].x;
            Bt[buf][c + 1][n] = (__bf16)wreg[i].y;
            Bt[buf][c + 2][n] = (__bf16)wreg[i].z;
            Bt[buf][c + 3][n] = (__bf16)wreg[i].w;
        }
    };

    // ---- prologue: stage K-step 0 into buffer 0 ----
    stageA(0, 0);
    loadW(0);
    storeW(0);
    wait_async0();
    __syncthreads();

    const int nk = K >> 5;
    for (int ks = 0; ks < nk; ++ks) {
        const int cur = ks & 1, nxt = cur ^ 1;
        const bool more = (ks + 1 < nk);

        // issue next tile's async A copies + W global loads (latency hidden
        // behind this step's WMMAs)
        if (more) { stageA(ks + 1, nxt); loadW(ks + 1); }

        // B fragments once (shared by both M sub-tiles)
        v16bf bfrag[4];
        #pragma unroll
        for (int t = 0; t < 4; ++t) {
            #pragma unroll
            for (int e = 0; e < 8; ++e) {
                bfrag[t][e]     = Bt[cur][koff + e][t * 16 + l15];
                bfrag[t][e + 8] = Bt[cur][koff + 16 + e][t * 16 + l15];
            }
        }
        #pragma unroll
        for (int u = 0; u < 2; ++u) {
            v8bf alo = *(const v8bf*)&As[cur][wave * 32 + u * 16 + l15][koff];
            v8bf ahi = *(const v8bf*)&As[cur][wave * 32 + u * 16 + l15][koff + 16];
            v16bf afrag = cat16(alo, ahi);
            #pragma unroll
            for (int t = 0; t < 4; ++t) {
                acc[u][t] = __builtin_amdgcn_wmma_f32_16x16x32_bf16(
                    false, afrag, false, bfrag[t], (short)0, acc[u][t], false, false);
            }
        }

        // convert + store next W slab, then close the stage
        if (more) storeW(nxt);
        wait_async0();
        __syncthreads();
    }

    // ---- epilogue ----
    #pragma unroll
    for (int u = 0; u < 2; ++u) {
        #pragma unroll
        for (int t = 0; t < 4; ++t) {
            #pragma unroll
            for (int r = 0; r < 8; ++r) {
                int m = mb + wave * 32 + u * 16 + r + half * 8;
                int n = nb + t * 16 + l15;
                size_t o = (size_t)m * N + n;
                float c = acc[u][t][r];
                if (bias) c += bias[n];
                if (GELU) c = gelu_tanh(c);
                if (res0) c += res0[o];
                if (res1) c += res1[o];
                if (Cf) Cf[o] = c;
                if (Cb) Cb[o] = (__bf16)c;
            }
        }
    }
}

// ---------------------------------------------------------------------------
// QKV rearrange + rotary (bf16 in, bf16 out).
// qkv[s][3072] cols per MP group p: [q(4x64) | v(4x64) | k(4x64)], h = p*4+j.
// Outputs: q[h][s][d], v[h][s][d] row-major; k stored transposed kt[h][d][s]
// so attention K staging is a contiguous copy.
// ---------------------------------------------------------------------------
__global__ __launch_bounds__(256) void rotary_split_kernel(const __bf16* __restrict__ qkv,
                                                           const int* __restrict__ pos,
                                                           __bf16* __restrict__ q,
                                                           __bf16* __restrict__ kt,
                                                           __bf16* __restrict__ v) {
    int idx = blockIdx.x * 256 + threadIdx.x;   // s*H*HD + h*HD + d
    int d = idx & (HDIM - 1);
    int h = (idx >> 6) & (NHEAD - 1);
    int s = idx >> 10;
    int p = h >> 2, j = h & 3;
    int base = s * 3 * DMODEL + p * 768 + j * 64;

    float vv = (float)qkv[base + 256 + d];
    float qv = (float)qkv[base + d];
    float kv = (float)qkv[base + 512 + d];

    if (d < ROTD) {
        int i = d >> 1;
        float inv = powf(10000.0f, -(float)(2 * i) / (float)ROTD);
        float ang = (float)pos[s] * inv;
        float sn = sinf(ang), cs = cosf(ang);
        float qp, kp;
        if (d & 1) { qp =  (float)qkv[base + d - 1];  kp =  (float)qkv[base + 512 + d - 1]; }
        else       { qp = -(float)qkv[base + d + 1];  kp = -(float)qkv[base + 512 + d + 1]; }
        qv = qv * cs + qp * sn;
        kv = kv * cs + kp * sn;
    }
    q [((size_t)h * SEQ + s) * HDIM + d] = (__bf16)qv;
    v [((size_t)h * SEQ + s) * HDIM + d] = (__bf16)vv;
    kt[((size_t)h * HDIM + d) * SEQ + s] = (__bf16)kv;
}

// ---------------------------------------------------------------------------
// Flash attention (causal). One block per (128-row q-tile, head).
// All Q/K/V staging is async global->LDS bf16 copy (no conversion, K already
// transposed in memory). Scores and P@V via WMMA; online softmax with
// half-wave shuffles. Output y is bf16 (operand of the Wo GEMM).
// ---------------------------------------------------------------------------
__global__ __launch_bounds__(256) void flash_attn_kernel(const __bf16* __restrict__ q,
                                                         const __bf16* __restrict__ kt,
                                                         const __bf16* __restrict__ v,
                                                         __bf16* __restrict__ y) {
    __shared__ __bf16 Qs[128][HDIM];       // 16 KB
    __shared__ __bf16 Kt[HDIM][64];        //  8 KB  [d][n]
    __shared__ __bf16 Vs[64][HDIM];        //  8 KB  [j][d]
    __shared__ __bf16 Ps[8][16][64];       // 16 KB  per-wave P tiles

    const int qt = blockIdx.x;             // 0..7
    const int h  = blockIdx.y;             // 0..15
    const int qb = qt * 128;

    const int tid  = threadIdx.x;
    const int lane = tid & 31;
    const int wave = tid >> 5;
    const int half = lane >> 4;
    const int l15  = lane & 15;
    const int koff = half ? 8 : 0;

    const __bf16* qh = q  + (size_t)h * SEQ * HDIM;
    const __bf16* kh = kt + (size_t)h * HDIM * SEQ;
    const __bf16* vh = v  + (size_t)h * SEQ * HDIM;

    // stage Q tile once (128x64 bf16 = 1024 x 16B)
    #pragma unroll
    for (int i = 0; i < 4; ++i) {
        int lin = tid + i * 256;
        int r = lin >> 3;
        int c = (lin & 7) * 8;
        async_copy_b128(&Qs[r][c], &qh[(size_t)(qb + r) * HDIM + c]);
    }

    float mstate[8], lstate[8];
    v8f o[4];
    #pragma unroll
    for (int r = 0; r < 8; ++r) { mstate[r] = -3.0e38f; lstate[r] = 0.0f; }
    #pragma unroll
    for (int t = 0; t < 4; ++t)
        #pragma unroll
        for (int e = 0; e < 8; ++e) o[t][e] = 0.0f;

    const int nkb = 2 * qt + 2;            // causal: keys up to qb+127
    for (int kb = 0; kb < nkb; ++kb) {
        // ---- stage K (already d-major) and V blocks: async copies ----
        #pragma unroll
        for (int i = 0; i < 2; ++i) {
            int lin = tid + i * 256;
            int d = lin >> 3;
            int c = (lin & 7) * 8;
            async_copy_b128(&Kt[d][c], &kh[(size_t)d * SEQ + kb * 64 + c]);
            async_copy_b128(&Vs[d][c], &vh[(size_t)(kb * 64 + d) * HDIM + c]);
        }
        wait_async0();
        __syncthreads();

        // ---- scores: S = Q @ K^T  (K-dim = HDIM = 64, two WMMA steps) ----
        v8f sc[4];
        #pragma unroll
        for (int t = 0; t < 4; ++t)
            #pragma unroll
            for (int e = 0; e < 8; ++e) sc[t][e] = 0.0f;

        #pragma unroll
        for (int d0 = 0; d0 < HDIM; d0 += 32) {
            v8bf alo = *(const v8bf*)&Qs[wave * 16 + l15][d0 + koff];
            v8bf ahi = *(const v8bf*)&Qs[wave * 16 + l15][d0 + koff + 16];
            v16bf afrag = cat16(alo, ahi);
            #pragma unroll
            for (int t = 0; t < 4; ++t) {
                v16bf bfrag;
                #pragma unroll
                for (int e = 0; e < 8; ++e) {
                    bfrag[e]     = Kt[d0 + koff + e][t * 16 + l15];
                    bfrag[e + 8] = Kt[d0 + koff + 16 + e][t * 16 + l15];
                }
                sc[t] = __builtin_amdgcn_wmma_f32_16x16x32_bf16(
                    false, afrag, false, bfrag, (short)0, sc[t], false, false);
            }
        }

        // ---- scale + causal mask ----
        #pragma unroll
        for (int t = 0; t < 4; ++t) {
            #pragma unroll
            for (int r = 0; r < 8; ++r) {
                int m = qb + wave * 16 + r + half * 8;
                int n = kb * 64 + t * 16 + l15;
                float val = sc[t][r] * 0.125f;  // 1/sqrt(64)
                sc[t][r] = (n > m) ? -3.0e38f : val;
            }
        }

        // ---- online softmax per row (rows live in 16-lane halves) ----
        #pragma unroll
        for (int r = 0; r < 8; ++r) {
            float mx = fmaxf(fmaxf(sc[0][r], sc[1][r]), fmaxf(sc[2][r], sc[3][r]));
            #pragma unroll
            for (int xm = 1; xm < 16; xm <<= 1)
                mx = fmaxf(mx, __shfl_xor(mx, xm, 32));
            float mnew = fmaxf(mstate[r], mx);
            float corr = __expf(mstate[r] - mnew);
            mstate[r] = mnew;
            float rsum = 0.0f;
            #pragma unroll
            for (int t = 0; t < 4; ++t) {
                float p = __expf(sc[t][r] - mnew);
                Ps[wave][r + half * 8][t * 16 + l15] = (__bf16)p;
                rsum += p;
            }
            #pragma unroll
            for (int xm = 1; xm < 16; xm <<= 1)
                rsum += __shfl_xor(rsum, xm, 32);
            lstate[r] = lstate[r] * corr + rsum;
            #pragma unroll
            for (int t = 0; t < 4; ++t) o[t][r] = o[t][r] * corr;
        }
        __syncthreads();

        // ---- O += P @ V  (K-dim = 64 keys, two WMMA steps) ----
        #pragma unroll
        for (int j0 = 0; j0 < 64; j0 += 32) {
            v8bf alo = *(const v8bf*)&Ps[wave][l15][j0 + koff];
            v8bf ahi = *(const v8bf*)&Ps[wave][l15][j0 + koff + 16];
            v16bf afrag = cat16(alo, ahi);
            #pragma unroll
            for (int t = 0; t < 4; ++t) {
                v16bf bfrag;
                #pragma unroll
                for (int e = 0; e < 8; ++e) {
                    bfrag[e]     = Vs[j0 + koff + e][t * 16 + l15];
                    bfrag[e + 8] = Vs[j0 + koff + 16 + e][t * 16 + l15];
                }
                o[t] = __builtin_amdgcn_wmma_f32_16x16x32_bf16(
                    false, afrag, false, bfrag, (short)0, o[t], false, false);
            }
        }
        __syncthreads();
    }

    // ---- normalize + write y[s][h*64+d] (bf16) ----
    #pragma unroll
    for (int t = 0; t < 4; ++t) {
        #pragma unroll
        for (int r = 0; r < 8; ++r) {
            int m = qb + wave * 16 + r + half * 8;
            int d = t * 16 + l15;
            y[(size_t)m * DMODEL + h * HDIM + d] = (__bf16)(o[t][r] / lstate[r]);
        }
    }
}

// ---------------------------------------------------------------------------
// Host-side orchestration
// ---------------------------------------------------------------------------
extern "C" void kernel_launch(void* const* d_in, const int* in_sizes, int n_in,
                              void* d_out, int out_size, void* d_ws, size_t ws_size,
                              hipStream_t stream) {
    const int*   idx  = (const int*)d_in[0];
    const int*   pos  = (const int*)d_in[1];
    const float* emb  = (const float*)d_in[2];
    const float* Wqkv = (const float*)d_in[3];
    const float* Wo   = (const float*)d_in[4];
    const float* ln1w = (const float*)d_in[5];
    const float* ln1b = (const float*)d_in[6];
    const float* fciw = (const float*)d_in[7];
    const float* fcib = (const float*)d_in[8];
    const float* fcow = (const float*)d_in[9];
    const float* fcob = (const float*)d_in[10];
    const float* lnfw = (const float*)d_in[11];
    const float* lnfb = (const float*)d_in[12];
    const float* lmw  = (const float*)d_in[13];
    const float* lmb  = (const float*)d_in[14];
    float* out = (float*)d_out;

    // workspace carving (256B-aligned chunks)
    char* p = (char*)d_ws;
    const size_t SD = (size_t)SEQ * DMODEL;
    float*  x        = (float*)p;            p += SD * 4;              // 4 MB
    float*  attn_out = (float*)p;            p += SD * 4;              // 4 MB
    __bf16* hbuf     = (__bf16*)p;           p += SD * 2;              // 2 MB
    __bf16* qkv      = (__bf16*)p;           p += 3 * SD * 2;          // 6 MB
    __bf16* qr       = (__bf16*)p;           p += SD * 2;              // 2 MB
    __bf16* ktr      = (__bf16*)p;           p += SD * 2;              // 2 MB
    __bf16* vr       = (__bf16*)p;           p += SD * 2;              // 2 MB
    __bf16* yb       = (__bf16*)p;           p += SD * 2;              // 2 MB
    __bf16* mid      = (__bf16*)p;           p += (size_t)SEQ * FFDIM * 2; // 8 MB

    dim3 blk(256);

    embed_kernel<<<dim3(SEQ), blk, 0, stream>>>(idx, emb, x);

    for (int l = 0; l < NLAYER; ++l) {
        const float* wqkv_l = Wqkv + (size_t)l * 3 * DMODEL * DMODEL;
        const float* wo_l   = Wo   + (size_t)l * DMODEL * DMODEL;
        const float* fiw_l  = fciw + (size_t)l * FFDIM * DMODEL;
        const float* fib_l  = fcib + (size_t)l * FFDIM;
        const float* fow_l  = fcow + (size_t)l * DMODEL * FFDIM;
        const float* fob_l  = fcob + (size_t)l * DMODEL;

        // h = LN(x)  (bf16)
        layernorm_kernel<<<dim3(SEQ), blk, 0, stream>>>(x, ln1w + l * DMODEL,
                                                        ln1b + l * DMODEL, hbuf);
        // qkv = h @ Wqkv^T  (bf16 out)
        gemm_wmma<false><<<dim3(3 * DMODEL / 64, SEQ / 256), blk, 0, stream>>>(
            hbuf, wqkv_l, nullptr, nullptr, nullptr, nullptr, qkv,
            SEQ, 3 * DMODEL, DMODEL);
        // rotary + head split (k transposed)
        rotary_split_kernel<<<dim3(SEQ * NHEAD * HDIM / 256), blk, 0, stream>>>(
            qkv, pos, qr, ktr, vr);
        // attention -> y (bf16)
        flash_attn_kernel<<<dim3(SEQ / 128, NHEAD), blk, 0, stream>>>(qr, ktr, vr, yb);
        // attn_out = y @ Wo^T  (fp32)
        gemm_wmma<false><<<dim3(DMODEL / 64, SEQ / 256), blk, 0, stream>>>(
            yb, wo_l, nullptr, nullptr, nullptr, attn_out, nullptr,
            SEQ, DMODEL, DMODEL);
        // mid = gelu(h @ fc_in^T + b)  (bf16 out)
        gemm_wmma<true><<<dim3(FFDIM / 64, SEQ / 256), blk, 0, stream>>>(
            hbuf, fiw_l, fib_l, nullptr, nullptr, nullptr, mid,
            SEQ, FFDIM, DMODEL);
        // x = (mid @ fc_out^T + b) + attn_out + x   (in-place residual, fp32)
        gemm_wmma<false><<<dim3(DMODEL / 64, SEQ / 256), blk, 0, stream>>>(
            mid, fow_l, fob_l, attn_out, x, x, nullptr,
            SEQ, DMODEL, FFDIM);
    }

    // final LN + LM head -> logits (fp32 to d_out)
    layernorm_kernel<<<dim3(SEQ), blk, 0, stream>>>(x, lnfw, lnfb, hbuf);
    gemm_wmma<false><<<dim3(VOCAB / 64, SEQ / 256), blk, 0, stream>>>(
        hbuf, lmw, lmb, nullptr, nullptr, out, nullptr,
        SEQ, VOCAB, DMODEL);
}
